// Attention_11132555231665
// MI455X (gfx1250) — matrix operand order repaired and checked by reading, not compiled, not run on hardware
//
#include <hip/hip_runtime.h>

typedef __attribute__((ext_vector_type(16))) _Float16 v16h;
typedef __attribute__((ext_vector_type(8)))  _Float16 v8h;
typedef __attribute__((ext_vector_type(4)))  _Float16 v4h;
typedef __attribute__((ext_vector_type(8)))  float    v8f;
typedef __attribute__((ext_vector_type(4)))  float    v4f;

#define S_LEN   2048
#define HEAD_D  64
#define Q_BLK   128     // queries per block (8 waves x 16)
#define K_TILE  64      // keys/values per LDS tile
#define LDS_STR 72      // halves per LDS row (64 + 8 pad) -> 144B: 16B-aligned, bank-friendly

// D = A(16x32 f16) x B(32x16 f16) + C(16x16 f32)
#define WMMA_F16(A, B, C) \
  __builtin_amdgcn_wmma_f32_16x16x32_f16(false, (A), false, (B), (short)0, (C), false, false)

// A-fragment (16x32, f16): lane m = row, halves e -> col = colBase + e + hi*16.
// Two contiguous 8-half runs -> two ds_load_b128.
__device__ __forceinline__ v16h lds_afrag(const _Float16* base, int row, int colBase, int hi) {
  const _Float16* p = base + row * LDS_STR + colBase + hi * 16;
  v8h lo = *(const v8h*)(p);
  v8h hh = *(const v8h*)(p + 8);
  return __builtin_shufflevector(lo, hh, 0,1,2,3,4,5,6,7,8,9,10,11,12,13,14,15);
}

// Build a B-fragment (32x16 f16, lane n holds K = e + hi*16) from two 16x16 f32 D-tiles
// a (rows 0..15 of K) and b (rows 16..31). D layout: lane lo holds M=v, lane hi M=v+8,
// so halves 8..15 (lo) / 0..7 (hi) come from the partner lane via xor-16 shuffle.
__device__ __forceinline__ v16h packBT(v8f a, v8f b, int hi) {
  v16h r;
#pragma unroll
  for (int e = 0; e < 8; ++e) {
    float ae = a[e], be = b[e];
    float sa = __shfl_xor(ae, 16, 32);
    float sb = __shfl_xor(be, 16, 32);
    r[e]     = (_Float16)(hi ? sb : ae);
    r[e + 8] = (_Float16)(hi ? be : sa);
  }
  return r;
}

__global__ __launch_bounds__(256)
void fa_fused_wmma_kernel(const float* __restrict__ Q, const float* __restrict__ K,
                          const float* __restrict__ V, const float* __restrict__ W,
                          const float* __restrict__ bias, float* __restrict__ out) {
  __shared__ __align__(16) _Float16 Klds [K_TILE * LDS_STR];  // [key][dim]   f16
  __shared__ __align__(16) _Float16 VTlds[HEAD_D * LDS_STR];  // [dim][key]   f16 (transposed)
  __shared__ __align__(16) _Float16 Wlds [HEAD_D * LDS_STR];  // [e_out][d]   f16

  const int nqblk = S_LEN / Q_BLK;
  const int bh    = blockIdx.x / nqblk;
  const int qblk  = blockIdx.x % nqblk;
  const size_t base = (size_t)bh * S_LEN * HEAD_D;

  const int t    = threadIdx.x;
  const int lane = t & 31;
  const int wv   = t >> 5;
  const int r    = lane & 15;
  const int hi   = lane >> 4;

  // ---- stage W_out once (64x64 f32 -> f16 LDS row-major) ----
  {
    const int sr = t >> 2, sc = (t & 3) * 16;
    const float* wg = W + sr * HEAD_D + sc;
#pragma unroll
    for (int j = 0; j < 4; ++j) {
      v4f w4 = *(const v4f*)(wg + j * 4);
      v4h h4;
#pragma unroll
      for (int i = 0; i < 4; ++i) h4[i] = (_Float16)w4[i];
      *(v4h*)(&Wlds[sr * LDS_STR + sc + j * 4]) = h4;
    }
  }

  // ---- load this wave's Q as B-fragments of Q^T (fold scale * log2(e)) ----
  const float qscale = 0.125f * 1.44269504088896340736f;  // 1/sqrt(64) * log2(e)
  const int qrow = qblk * Q_BLK + wv * 16 + r;
  v16h qf[2];
  {
    const float* qp = Q + base + (size_t)qrow * HEAD_D;
#pragma unroll
    for (int c = 0; c < 2; ++c) {
      const float* p = qp + c * 32 + hi * 16;   // lane holds 16 consecutive dims
#pragma unroll
      for (int e = 0; e < 16; ++e) qf[c][e] = (_Float16)(p[e] * qscale);
    }
  }

  v8f accO[4];                 // O^T accumulator: 4 tiles of (16 dims x 16 queries), f32
#pragma unroll
  for (int c = 0; c < 4; ++c) accO[c] = (v8f){};
  float mrow = -INFINITY, lrow = 0.f;   // per-query running max (log2 domain) and sum

  for (int kt = 0; kt < S_LEN; kt += K_TILE) {
    __syncthreads();  // previous tile's LDS reads complete
    // ---- stage K tile (row-major) and V tile (transposed) into LDS as f16 ----
    {
      const int sr = t >> 2, sc = (t & 3) * 16;
      const float* kg = K + base + (size_t)(kt + sr) * HEAD_D + sc;
      const float* vg = V + base + (size_t)(kt + sr) * HEAD_D + sc;
#pragma unroll
      for (int j = 0; j < 4; ++j) {
        v4f k4 = *(const v4f*)(kg + j * 4);
        v4h h4;
#pragma unroll
        for (int i = 0; i < 4; ++i) h4[i] = (_Float16)k4[i];
        *(v4h*)(&Klds[sr * LDS_STR + sc + j * 4]) = h4;
        v4f v4 = *(const v4f*)(vg + j * 4);
#pragma unroll
        for (int i = 0; i < 4; ++i)
          VTlds[(sc + j * 4 + i) * LDS_STR + sr] = (_Float16)v4[i];
      }
      if (kt + K_TILE < S_LEN) {      // gfx1250 global_prefetch_b8 for next tile
        __builtin_prefetch(kg + K_TILE * HEAD_D, 0, 1);
        __builtin_prefetch(vg + K_TILE * HEAD_D, 0, 1);
      }
    }
    __syncthreads();

    // ---- S^T = K . Q^T : 4 groups of 16 keys, accumulate over two 32-dim chunks ----
    v8f st[4];
#pragma unroll
    for (int g = 0; g < 4; ++g) {
      v16h k0 = lds_afrag(Klds, g * 16 + r, 0,  hi);
      v16h k1 = lds_afrag(Klds, g * 16 + r, 32, hi);
      v8f acc = (v8f){};
      acc = WMMA_F16(k0, qf[0], acc);
      acc = WMMA_F16(k1, qf[1], acc);
      st[g] = acc;   // lane = query, vgpr v = key 16g+v (+8 for hi lanes)
    }

    // ---- online softmax (exp2 domain; keys live in the per-lane vgpr dim) ----
    float mp = -INFINITY;
#pragma unroll
    for (int g = 0; g < 4; ++g)
#pragma unroll
      for (int v = 0; v < 8; ++v) mp = fmaxf(mp, st[g][v]);
    mp = fmaxf(mp, __shfl_xor(mp, 16, 32));
    const float mnew  = fmaxf(mrow, mp);
    const float alpha = exp2f(mrow - mnew);
    float ssum = 0.f;
#pragma unroll
    for (int g = 0; g < 4; ++g)
#pragma unroll
      for (int v = 0; v < 8; ++v) {
        float e = exp2f(st[g][v] - mnew);
        st[g][v] = e;
        ssum += e;
      }
    ssum += __shfl_xor(ssum, 16, 32);
    lrow = lrow * alpha + ssum;
    mrow = mnew;
#pragma unroll
    for (int c = 0; c < 4; ++c) accO[c] *= alpha;

    // ---- O^T += V^T . P^T ----
    v16h p0 = packBT(st[0], st[1], hi);   // keys kt+0..31
    v16h p1 = packBT(st[2], st[3], hi);   // keys kt+32..63
#pragma unroll
    for (int c = 0; c < 4; ++c) {
      v16h vf0 = lds_afrag(VTlds, c * 16 + r, 0,  hi);
      v16h vf1 = lds_afrag(VTlds, c * 16 + r, 32, hi);
      accO[c] = WMMA_F16(vf0, p0, accO[c]);
      accO[c] = WMMA_F16(vf1, p1, accO[c]);
    }
  }

  // ---- normalize, then fused projection: final^T = W . O^T, + bias ----
  const float inv_l = 1.0f / lrow;
#pragma unroll
  for (int c = 0; c < 4; ++c) accO[c] *= inv_l;
  v16h of0 = packBT(accO[0], accO[1], hi);   // dims 0..31  as B-fragment
  v16h of1 = packBT(accO[2], accO[3], hi);   // dims 32..63

  float* orow = out + base + (size_t)qrow * HEAD_D;
#pragma unroll
  for (int mc = 0; mc < 4; ++mc) {
    v16h w0 = lds_afrag(Wlds, mc * 16 + r, 0,  hi);
    v16h w1 = lds_afrag(Wlds, mc * 16 + r, 32, hi);
    v8f f = (v8f){};
    f = WMMA_F16(w0, of0, f);
    f = WMMA_F16(w1, of1, f);
    v4f o0, o1;
#pragma unroll
    for (int v = 0; v < 4; ++v) {
      o0[v] = f[v]     + bias[mc * 16 + hi * 8 + v];
      o1[v] = f[v + 4] + bias[mc * 16 + hi * 8 + 4 + v];
    }
    *(v4f*)(orow + mc * 16 + hi * 8)     = o0;
    *(v4f*)(orow + mc * 16 + hi * 8 + 4) = o1;
  }
}

extern "C" void kernel_launch(void* const* d_in, const int* in_sizes, int n_in,
                              void* d_out, int out_size, void* d_ws, size_t ws_size,
                              hipStream_t stream) {
  (void)n_in; (void)out_size; (void)d_ws; (void)ws_size;
  const float* Q    = (const float*)d_in[0];
  const float* K    = (const float*)d_in[1];
  const float* V    = (const float*)d_in[2];
  const float* W    = (const float*)d_in[3];
  const float* bias = (const float*)d_in[4];
  float* out = (float*)d_out;

  const int nbh = in_sizes[0] / (S_LEN * HEAD_D);        // B*H = 64
  dim3 grid(nbh * (S_LEN / Q_BLK));                      // 64 * 16 = 1024 blocks
  fa_fused_wmma_kernel<<<grid, 256, 0, stream>>>(Q, K, V, W, bias, out);
}
